// OneDParallelMHA_77850577207467
// MI455X (gfx1250) — matrix operand order repaired and checked by reading, not compiled, not run on hardware
//
#include <hip/hip_runtime.h>

typedef __attribute__((ext_vector_type(16))) _Float16 v16h;
typedef __attribute__((ext_vector_type(8)))  _Float16 v8h;
typedef __attribute__((ext_vector_type(8)))  float    v8f;
typedef __attribute__((ext_vector_type(4)))  unsigned int u32x4;
typedef __attribute__((ext_vector_type(8)))  unsigned int u32x8;

#define WMMA_F16(a, b, c) \
  __builtin_amdgcn_wmma_f32_16x16x32_f16(false, (a), false, (b), (short)0, (c), false, false)

static constexpr int Bsz = 2, Ssz = 2048, Esz = 2048, Hn = 16, Dh = 128;
static constexpr int Mrows = Bsz * Ssz;      // 4096
static constexpr int F3E   = 3 * Esz;        // 6144

union HU { v16h v; v8h h[2]; };

__device__ __forceinline__ v16h load2x8(const _Float16* p0, const _Float16* p1) {
  HU u;
  u.h[0] = *(const v8h*)p0;
  u.h[1] = *(const v8h*)p1;
  return u.v;
}

// ---------------------------------------------------------------- fp32 -> f16
__global__ void cvt_f32_f16(const float* __restrict__ src, _Float16* __restrict__ dst, int n) {
  int i = blockIdx.x * blockDim.x + threadIdx.x;
  if (i < n) dst[i] = (_Float16)src[i];
}

// ------------------------------------------------- C = A[M,K] * B[N,K]^T + bias
// Wave tile: 32x64 (2 A frags x 4 B frags = 8 wmma/step). Block: 4 waves -> 128x64.
template <bool OUT_F32>
__global__ void __launch_bounds__(128)
wmma_gemm_nt(const _Float16* __restrict__ A, const _Float16* __restrict__ Bm,
             const float* __restrict__ bias, void* __restrict__ Cp,
             int M, int N, int K) {
  const int lane = threadIdx.x & 31;
  const int wave = threadIdx.x >> 5;
  const int hh = lane >> 4, lm = lane & 15;
  const int m0 = blockIdx.x * 128 + wave * 32;
  const int n0 = blockIdx.y * 64;

  v8f acc[2][4] = {};
  const _Float16* arow0 = A + (size_t)(m0 + lm) * (size_t)K;
  const _Float16* arow1 = arow0 + (size_t)16 * (size_t)K;

  for (int k = 0; k < K; k += 32) {
    v16h a0 = load2x8(arow0 + k + 16 * hh, arow0 + k + 16 * hh + 8);
    v16h a1 = load2x8(arow1 + k + 16 * hh, arow1 + k + 16 * hh + 8);
#pragma unroll
    for (int nt = 0; nt < 4; ++nt) {
      const _Float16* brow = Bm + (size_t)(n0 + 16 * nt + lm) * (size_t)K + k + 16 * hh;
      v16h b = load2x8(brow, brow + 8);
      acc[0][nt] = WMMA_F16(a0, b, acc[0][nt]);
      acc[1][nt] = WMMA_F16(a1, b, acc[1][nt]);
    }
  }

#pragma unroll
  for (int mi = 0; mi < 2; ++mi) {
#pragma unroll
    for (int nt = 0; nt < 4; ++nt) {
#pragma unroll
      for (int r = 0; r < 8; ++r) {
        int row = m0 + 16 * mi + r + 8 * hh;
        int col = n0 + 16 * nt + lm;
        float v = acc[mi][nt][r] + bias[col];
        if (OUT_F32)
          ((float*)Cp)[(size_t)row * (size_t)N + col] = v;
        else
          ((_Float16*)Cp)[(size_t)row * (size_t)N + col] = (_Float16)v;
      }
    }
  }
}

// ------------------------------------------- RoPE + split qkv into Q/K/V layouts
__global__ void rope_split(const _Float16* __restrict__ qkv, _Float16* __restrict__ Q,
                           _Float16* __restrict__ Kq, _Float16* __restrict__ V) {
  int idx = blockIdx.x * blockDim.x + threadIdx.x;   // B*H*S*64 threads
  int d = idx & 63;
  int s = (idx >> 6) & (Ssz - 1);
  int h = (idx >> 17) & (Hn - 1);
  int b = idx >> 21;

  size_t mrow = ((size_t)b * Ssz + s) * (size_t)F3E;
  int base = h * Dh + d;

  float q1 = (float)qkv[mrow + base],            q2 = (float)qkv[mrow + base + 64];
  float k1 = (float)qkv[mrow + Esz + base],      k2 = (float)qkv[mrow + Esz + base + 64];
  float v1 = (float)qkv[mrow + 2 * Esz + base],  v2 = (float)qkv[mrow + 2 * Esz + base + 64];

  float inv_freq = __expf(-logf(10000.0f) * (float)(2 * d) * (1.0f / 128.0f));
  float ang = (float)s * inv_freq;
  float sn, cs;
  __sincosf(ang, &sn, &cs);

  size_t qo = (((size_t)b * Hn + h) * Ssz + s) * (size_t)Dh + d;
  Q[qo]       = (_Float16)(q1 * cs - q2 * sn);
  Q[qo + 64]  = (_Float16)(q1 * sn + q2 * cs);
  Kq[qo]      = (_Float16)(k1 * cs - k2 * sn);
  Kq[qo + 64] = (_Float16)(k1 * sn + k2 * cs);

  size_t vo = (((size_t)b * Hn + h) * (size_t)Dh + d) * (size_t)Ssz + s;
  V[vo]             = (_Float16)v1;
  V[vo + 64 * Ssz]  = (_Float16)v2;
}

// ---------------------------------------------------- causal flash attention
// One wave per (b,h, 16-row q tile). 32 keys/step: TDM stages the 32x128 K tile
// into LDS (padded rows: 256B + 16B -> bank-conflict-free), 8 QK wmma + 8 PV wmma.
__global__ void __launch_bounds__(32)
flash_attn_wmma(const _Float16* __restrict__ Q, const _Float16* __restrict__ Km,
                const _Float16* __restrict__ V, _Float16* __restrict__ ctx) {
  __shared__ _Float16 klds[32][136];   // 128 halfs + 8 pad (TDM pad: 4 dw / 64 dw)
  __shared__ _Float16 plds[16][40];    // P tile, padded row stride

  const int lane = threadIdx.x & 31;
  const int hh = lane >> 4, lm = lane & 15;
  const int qt = blockIdx.x;
  const int bh = blockIdx.y;
  const int b = bh >> 4, h = bh & 15;
  const int q0 = qt * 16;

  const _Float16* Qb = Q + (size_t)bh * Ssz * Dh;
  const _Float16* Kb = Km + (size_t)bh * Ssz * Dh;
  const _Float16* Vb = V + (size_t)bh * Dh * Ssz;

  // TDM descriptor group1 (constant): data_size=8B, pad_enable, pad_interval=64dw,
  // pad_amount=4dw, tensor/tile = 32x32 (u64 units), dim0 stride = 32.
  const u32x8 g1 = {0x07530000u, 0x00200000u, 0x00200000u, 0x00200000u,
                    32u, 32u, 0u, 0u};
  const unsigned klds_addr = (unsigned)(uintptr_t)&klds[0][0];

  // Q fragments for K-dim = 128 (4 x 16x32)
  v16h qa[4];
  const _Float16* qrow = Qb + (size_t)(q0 + lm) * Dh;
#pragma unroll
  for (int j = 0; j < 4; ++j)
    qa[j] = load2x8(qrow + 32 * j + 16 * hh, qrow + 32 * j + 16 * hh + 8);

  v8f cacc[8] = {};
  float rmax[8], rsum[8];
#pragma unroll
  for (int r = 0; r < 8; ++r) { rmax[r] = -1e30f; rsum[r] = 0.0f; }

  const float scale = 0.08838834764831845f;  // 1/sqrt(128)

  for (int k0 = 0; k0 < q0 + 16; k0 += 32) {
    // --- async TDM load: K rows [k0, k0+32) (32 x 256B) -> klds ---
    unsigned long long gaddr = (unsigned long long)(uintptr_t)(Kb + (size_t)k0 * Dh);
    u32x4 g0;
    g0.x = 1u;                                        // count=1, user descriptor
    g0.y = klds_addr;                                 // lds_addr
    g0.z = (unsigned)(gaddr & 0xffffffffu);           // global_addr[31:0]
    g0.w = (unsigned)((gaddr >> 32) & 0x01ffffffu) | 0x80000000u;  // [56:32] | type=2
    asm volatile("tensor_load_to_lds %0, %1" :: "s"(g0), "s"(g1) : "memory");
    __builtin_amdgcn_s_wait_tensorcnt(0);
    asm volatile("" ::: "memory");

    // S = Q * K^T for 32 keys (two 16-key column halves), K from LDS
    v8f sf[2] = {};
#pragma unroll
    for (int f = 0; f < 2; ++f) {
      const _Float16* krow = &klds[16 * f + lm][16 * hh];
#pragma unroll
      for (int j = 0; j < 4; ++j) {
        v16h bfr = load2x8(krow + 32 * j, krow + 32 * j + 8);
        sf[f] = WMMA_F16(qa[j], bfr, sf[f]);
      }
    }

    // scale + causal mask
#pragma unroll
    for (int f = 0; f < 2; ++f) {
      int col = k0 + 16 * f + lm;
#pragma unroll
      for (int r = 0; r < 8; ++r) {
        int row = q0 + r + 8 * hh;
        float s = sf[f][r] * scale;
        sf[f][r] = (col <= row) ? s : -1e30f;
      }
    }

    // online softmax
    float mnew[8], corr[8];
#pragma unroll
    for (int r = 0; r < 8; ++r) {
      float v = fmaxf(sf[0][r], sf[1][r]);
      v = fmaxf(v, __shfl_xor(v, 1, 32));
      v = fmaxf(v, __shfl_xor(v, 2, 32));
      v = fmaxf(v, __shfl_xor(v, 4, 32));
      v = fmaxf(v, __shfl_xor(v, 8, 32));
      mnew[r] = fmaxf(rmax[r], v);
      corr[r] = __expf(rmax[r] - mnew[r]);
      rmax[r] = mnew[r];
    }

    float ps[8];
#pragma unroll
    for (int r = 0; r < 8; ++r) ps[r] = 0.0f;
#pragma unroll
    for (int f = 0; f < 2; ++f) {
#pragma unroll
      for (int r = 0; r < 8; ++r) {
        float p = __expf(sf[f][r] - mnew[r]);
        ps[r] += p;
        plds[r + 8 * hh][16 * f + lm] = (_Float16)p;   // C-layout -> (m, kcol)
      }
    }
#pragma unroll
    for (int r = 0; r < 8; ++r) {
      float v = ps[r];
      v += __shfl_xor(v, 1, 32);
      v += __shfl_xor(v, 2, 32);
      v += __shfl_xor(v, 4, 32);
      v += __shfl_xor(v, 8, 32);
      rsum[r] = rsum[r] * corr[r] + v;
#pragma unroll
      for (int c = 0; c < 8; ++c) cacc[c][r] *= corr[r];
    }
    __syncthreads();

    // reload P in A-fragment layout (row lm, K = i + 16*hh)
    v16h pa;
#pragma unroll
    for (int i = 0; i < 16; ++i)
      pa[i] = plds[lm][i + 16 * hh];
    __syncthreads();

    // ctx += P * V  (V stored [b,h,d,s]: B-frag contiguous along s)
#pragma unroll
    for (int c = 0; c < 8; ++c) {
      const _Float16* vrow = Vb + (size_t)(16 * c + lm) * Ssz + k0 + 16 * hh;
      v16h bv = load2x8(vrow, vrow + 8);
      cacc[c] = WMMA_F16(pa, bv, cacc[c]);
    }
  }

  // normalize and store ctx as [b, s, e] f16 (e = h*128 + d)
#pragma unroll
  for (int r = 0; r < 8; ++r) {
    float inv = 1.0f / rsum[r];
    int srow = q0 + r + 8 * hh;
#pragma unroll
    for (int c = 0; c < 8; ++c) {
      int e = h * Dh + 16 * c + lm;
      ctx[((size_t)b * Ssz + srow) * (size_t)Esz + e] = (_Float16)(cacc[c][r] * inv);
    }
  }
}

// ------------------------------------------------------------------- launcher
extern "C" void kernel_launch(void* const* d_in, const int* in_sizes, int n_in,
                              void* d_out, int out_size, void* d_ws, size_t ws_size,
                              hipStream_t stream) {
  const float* x      = (const float*)d_in[0];
  const float* wqkv_w = (const float*)d_in[1];
  const float* wqkv_b = (const float*)d_in[2];
  const float* out_w  = (const float*)d_in[3];
  const float* out_b  = (const float*)d_in[4];
  float* out = (float*)d_out;

  char* ws = (char*)d_ws;
  size_t off = 0;
  _Float16* xh    = (_Float16*)(ws + off); off += (size_t)Mrows * Esz * 2;
  _Float16* wqkvh = (_Float16*)(ws + off); off += (size_t)F3E * Esz * 2;
  _Float16* owh   = (_Float16*)(ws + off); off += (size_t)Esz * Esz * 2;
  _Float16* qkvh  = (_Float16*)(ws + off); off += (size_t)Mrows * F3E * 2;
  _Float16* qh    = (_Float16*)(ws + off); off += (size_t)Mrows * Esz * 2;
  _Float16* kh    = (_Float16*)(ws + off); off += (size_t)Mrows * Esz * 2;
  _Float16* vh    = (_Float16*)(ws + off); off += (size_t)Mrows * Esz * 2;
  _Float16* ctxh  = (_Float16*)(ws + off); off += (size_t)Mrows * Esz * 2;

  int nx  = Mrows * Esz;
  int nwq = F3E * Esz;
  int now = Esz * Esz;
  cvt_f32_f16<<<(nx  + 255) / 256, 256, 0, stream>>>(x, xh, nx);
  cvt_f32_f16<<<(nwq + 255) / 256, 256, 0, stream>>>(wqkv_w, wqkvh, nwq);
  cvt_f32_f16<<<(now + 255) / 256, 256, 0, stream>>>(out_w, owh, now);

  // qkv = x * Wqkv^T + b     (M=4096, N=6144, K=2048)
  wmma_gemm_nt<false><<<dim3(Mrows / 128, F3E / 64), 128, 0, stream>>>(
      xh, wqkvh, wqkv_b, (void*)qkvh, Mrows, F3E, Esz);

  int nrope = Bsz * Hn * Ssz * 64;
  rope_split<<<nrope / 256, 256, 0, stream>>>(qkvh, qh, kh, vh);

  flash_attn_wmma<<<dim3(Ssz / 16, Bsz * Hn), 32, 0, stream>>>(qh, kh, vh, ctxh);

  // out = ctx * Wout^T + b   (M=4096, N=2048, K=2048), fp32 output
  wmma_gemm_nt<true><<<dim3(Mrows / 128, Esz / 64), 128, 0, stream>>>(
      ctxh, owh, out_b, (void*)out, Mrows, Esz, Esz);
}
